// MultiHeadAttention_88888643158578
// MI455X (gfx1250) — matrix operand order repaired and checked by reading, not compiled, not run on hardware
//
#include <hip/hip_runtime.h>
#include <hip/hip_bf16.h>

typedef __attribute__((ext_vector_type(16))) _Float16 v16h;
typedef __attribute__((ext_vector_type(8)))  _Float16 h8;
typedef __attribute__((ext_vector_type(8)))  float    v8f;
typedef __attribute__((ext_vector_type(4)))  float    f4;

#define B_   4
#define T_   2048
#define E_   1024
#define HS_  64
#define BT_  (B_ * T_)
#define PSTR 40   // padded LDS row stride (halfs) for the P staging tile

__device__ __forceinline__ v8f wmma_f16(v16h a, v16h b, v8f c) {
  return __builtin_amdgcn_wmma_f32_16x16x32_f16(false, a, false, b, (short)0, c, false, false);
}

// A-fragment (16x32 f16) loaded from an f32 row-major matrix with on-the-fly cvt.
__device__ __forceinline__ v16h wmma_a_from_f32(const float* __restrict__ base,
                                                int row0, int k0, int ld, int lane) {
  int r = row0 + (lane & 15);
  const float* p = base + (size_t)r * ld + k0 + ((lane >> 4) << 3);
  f4 x0 = *(const f4*)(p);
  f4 x1 = *(const f4*)(p + 4);
  f4 x2 = *(const f4*)(p + 16);
  f4 x3 = *(const f4*)(p + 20);
  v16h a;
  #pragma unroll
  for (int i = 0; i < 4; ++i) {
    a[i]      = (_Float16)x0[i];
    a[i + 4]  = (_Float16)x1[i];
    a[i + 8]  = (_Float16)x2[i];
    a[i + 12] = (_Float16)x3[i];
  }
  return a;
}

// A-fragment (16x32) from f16 row-major matrix, row stride ld (halfs).
__device__ __forceinline__ v16h wmma_a_f16(const _Float16* __restrict__ base,
                                           int row0, int k0, int ld, int lane) {
  int r = row0 + (lane & 15);
  const _Float16* p = base + (size_t)r * ld + k0 + ((lane >> 4) << 3);
  h8 lo = *(const h8*)(p);
  h8 hi = *(const h8*)(p + 16);
  v16h a;
  #pragma unroll
  for (int i = 0; i < 8; ++i) { a[i] = lo[i]; a[i + 8] = hi[i]; }
  return a;
}

// B-fragment (32x16): bt is B^T, i.e. bt[n][k]; lane reads halfs 8h..8h+7 and 16+8h..16+8h+7.
__device__ __forceinline__ v16h wmma_b_f16(const _Float16* __restrict__ bt,
                                           int n0, int k0, int ld, int lane) {
  const _Float16* p = bt + (size_t)(n0 + (lane & 15)) * ld + k0 + ((lane >> 4) << 3);
  h8 lo = *(const h8*)(p);
  h8 hi = *(const h8*)(p + 16);
  v16h b;
  #pragma unroll
  for (int i = 0; i < 8; ++i) { b[i] = lo[i]; b[i + 8] = hi[i]; }
  return b;
}

// ---- Kernel 1: weight prep -------------------------------------------------
// WqT/WkT/WvT[h][e] = W[e][h] (f16);  WpT[n][j] = sum_h Wp[h*64+j][n] (f16).
__global__ void __launch_bounds__(256)
prep_weights(const float* __restrict__ Wq, const float* __restrict__ Wk,
             const float* __restrict__ Wv, const float* __restrict__ Wp,
             _Float16* __restrict__ WqT, _Float16* __restrict__ WkT,
             _Float16* __restrict__ WvT, _Float16* __restrict__ WpT) {
  int idx = blockIdx.x * blockDim.x + threadIdx.x;   // 0..65535
  int h = idx >> 10, e = idx & 1023;                 // transpose mapping
  WqT[idx] = (_Float16)Wq[e * HS_ + h];
  WkT[idx] = (_Float16)Wk[e * HS_ + h];
  WvT[idx] = (_Float16)Wv[e * HS_ + h];
  int n = idx >> 6, j = idx & 63;                    // Wp fold mapping
  float s = 0.f;
  #pragma unroll
  for (int hh = 0; hh < 16; ++hh) s += Wp[(size_t)(hh * HS_ + j) * E_ + n];
  WpT[idx] = (_Float16)s;
}

// ---- Kernel 2: fused QKV projection (M=8192, N=64, K=1024) -----------------
// One wave per 16-row tile; 12 accumulators (q,k,v x 4 n-tiles); x read once.
__global__ void __launch_bounds__(256)
qkv_wmma(const float* __restrict__ x,
         const _Float16* __restrict__ WqT, const _Float16* __restrict__ WkT,
         const _Float16* __restrict__ WvT,
         _Float16* __restrict__ qh, _Float16* __restrict__ kh,
         _Float16* __restrict__ vt) {
  const int lane = threadIdx.x & 31;
  const int wave = threadIdx.x >> 5;
  const int row0 = (blockIdx.x * 8 + wave) * 16;     // 512 tiles total
  v8f aq[4] = {}, ak[4] = {}, av[4] = {};
  for (int k0 = 0; k0 < E_; k0 += 32) {
    v16h a = wmma_a_from_f32(x, row0, k0, E_, lane);
    #pragma unroll
    for (int nt = 0; nt < 4; ++nt) {
      v16h bq = wmma_b_f16(WqT, nt * 16, k0, E_, lane);
      aq[nt] = wmma_f16(a, bq, aq[nt]);
      v16h bk = wmma_b_f16(WkT, nt * 16, k0, E_, lane);
      ak[nt] = wmma_f16(a, bk, ak[nt]);
      v16h bv = wmma_b_f16(WvT, nt * 16, k0, E_, lane);
      av[nt] = wmma_f16(a, bv, av[nt]);
    }
  }
  const int base8 = (lane >> 4) << 3;
  const int col   = lane & 15;
  const int g0    = row0 + base8;       // first global row this lane owns
  const int bb    = g0 >> 11;           // batch
  const int t0    = g0 & 2047;          // position within batch (mult of 8)
  #pragma unroll
  for (int nt = 0; nt < 4; ++nt) {
    #pragma unroll
    for (int r = 0; r < 8; ++r) {
      // fold softmax scale hs^-0.5 = 0.125 into q
      qh[(size_t)(g0 + r) * HS_ + nt * 16 + col] = (_Float16)(aq[nt][r] * 0.125f);
      kh[(size_t)(g0 + r) * HS_ + nt * 16 + col] = (_Float16)(ak[nt][r]);
    }
    h8 pk;
    #pragma unroll
    for (int r = 0; r < 8; ++r) pk[r] = (_Float16)(av[nt][r]);
    // V stored transposed per batch: vt[b][h][t] -> contiguous keys for B-frags
    *(h8*)(vt + (((size_t)(bb * HS_ + nt * 16 + col)) << 11) + t0) = pk;
  }
}

// ---- Kernel 3: causal flash attention, 1 wave per 16-query tile ------------
__global__ void __launch_bounds__(128)
attn_wmma(const _Float16* __restrict__ qh, const _Float16* __restrict__ kh,
          const _Float16* __restrict__ vt, _Float16* __restrict__ headh) {
  __shared__ __align__(16) _Float16 ldsP[4][16 * PSTR];
  const int lane  = threadIdx.x & 31;
  const int wave  = threadIdx.x >> 5;
  _Float16* lp    = ldsP[wave];
  const int tile  = blockIdx.x * 4 + wave;   // 0..511
  const int q0g   = tile * 16;               // global query row
  const int b     = q0g >> 11;
  const int qt0   = q0g & 2047;              // query pos within batch
  const int base8 = (lane >> 4) << 3;
  const int col   = lane & 15;

  // Q tile resident (16x64 = two A-fragments), already scaled by 1/8
  v16h a0 = wmma_a_f16(qh, q0g, 0,  HS_, lane);
  v16h a1 = wmma_a_f16(qh, q0g, 32, HS_, lane);

  float mrow[8], lrow[8];
  v8f o[4] = {};
  #pragma unroll
  for (int r = 0; r < 8; ++r) { mrow[r] = -3.0e38f; lrow[r] = 0.f; }

  const _Float16* khb = kh + ((size_t)b << 11) * HS_;
  const _Float16* vtb = vt + ((size_t)b * HS_ << 11);
  const int nk = qt0 + 16;                   // keys needed (causal)
  for (int k0 = 0; k0 < nk; k0 += 32) {
    // S = Q @ K^T for 32 keys (2 n-tiles x 2 k-steps)
    v8f s0 = {}, s1 = {};
    v16h b00 = wmma_b_f16(khb, k0,      0,  HS_, lane);
    v16h b01 = wmma_b_f16(khb, k0,      32, HS_, lane);
    s0 = wmma_f16(a0, b00, s0); s0 = wmma_f16(a1, b01, s0);
    v16h b10 = wmma_b_f16(khb, k0 + 16, 0,  HS_, lane);
    v16h b11 = wmma_b_f16(khb, k0 + 16, 32, HS_, lane);
    s1 = wmma_f16(a0, b10, s1); s1 = wmma_f16(a1, b11, s1);

    if (k0 + 32 < nk)   // hint next K tile into cache
      __builtin_prefetch(khb + (size_t)(k0 + 32 + col) * HS_, 0, 1);

    if (k0 + 31 > qt0) {                     // diagonal tile: causal mask
      int c0 = k0 + col;
      #pragma unroll
      for (int r = 0; r < 8; ++r) {
        int qp = qt0 + base8 + r;
        if (c0 > qp)      s0[r] = -3.0e38f;
        if (c0 + 16 > qp) s1[r] = -3.0e38f;
      }
    }

    // online softmax; rows live on 16-lane halves -> xor-shuffle reductions
    #pragma unroll
    for (int r = 0; r < 8; ++r) {
      float v0 = s0[r], v1 = s1[r];
      float mx = fmaxf(v0, v1);
      #pragma unroll
      for (int off = 1; off < 16; off <<= 1)
        mx = fmaxf(mx, __shfl_xor(mx, off, 32));
      float mnew  = fmaxf(mrow[r], mx);
      float alpha = __expf(mrow[r] - mnew);
      float p0 = __expf(v0 - mnew);
      float p1 = __expf(v1 - mnew);
      float ps = p0 + p1;
      #pragma unroll
      for (int off = 1; off < 16; off <<= 1)
        ps += __shfl_xor(ps, off, 32);
      lrow[r] = lrow[r] * alpha + ps;
      mrow[r] = mnew;
      #pragma unroll
      for (int aa = 0; aa < 4; ++aa) o[aa][r] *= alpha;
      // stage P (C-layout) into LDS row-major for A-fragment reload
      lp[(base8 + r) * PSTR + col]      = (_Float16)p0;
      lp[(base8 + r) * PSTR + 16 + col] = (_Float16)p1;
    }

    // reload P as a 16x32 A-fragment (intra-wave LDS RAW; ds ops in-order)
    v16h ap;
    {
      const _Float16* p = lp + col * PSTR + ((lane >> 4) << 3);
      h8 lo = *(const h8*)p;
      h8 hi = *(const h8*)(p + 16);
      #pragma unroll
      for (int i = 0; i < 8; ++i) { ap[i] = lo[i]; ap[i + 8] = hi[i]; }
    }
    // O += P @ V (K=32 keys, 4 n-tiles of head dim)
    #pragma unroll
    for (int aa = 0; aa < 4; ++aa) {
      v16h bv = wmma_b_f16(vtb, aa * 16, k0, T_, lane);
      o[aa] = wmma_f16(ap, bv, o[aa]);
    }
  }

  #pragma unroll
  for (int aa = 0; aa < 4; ++aa)
    #pragma unroll
    for (int r = 0; r < 8; ++r)
      headh[(size_t)(q0g + base8 + r) * HS_ + aa * 16 + col] =
          (_Float16)(o[aa][r] / lrow[r]);
}

// ---- Kernel 4: out = head @ Wp_eff + bp  (M=8192, N=1024, K=64) ------------
__global__ void __launch_bounds__(256)
proj_wmma(const _Float16* __restrict__ headh, const _Float16* __restrict__ WpT,
          const float* __restrict__ bp, float* __restrict__ out) {
  const int lane = threadIdx.x & 31;
  const int wave = threadIdx.x >> 5;
  const int w    = blockIdx.x * 8 + wave;   // 0..8191
  const int row0 = (w >> 4) * 16;
  const int n0   = (w & 15) * 64;
  v8f c[4] = {};
  #pragma unroll
  for (int ks = 0; ks < 2; ++ks) {
    int k0 = ks * 32;
    v16h a = wmma_a_f16(headh, row0, k0, HS_, lane);
    #pragma unroll
    for (int nt = 0; nt < 4; ++nt) {
      v16h bm = wmma_b_f16(WpT, n0 + nt * 16, k0, HS_, lane);
      c[nt] = wmma_f16(a, bm, c[nt]);
    }
  }
  const int base8 = (lane >> 4) << 3;
  const int col   = lane & 15;
  #pragma unroll
  for (int nt = 0; nt < 4; ++nt) {
    float bias = bp[n0 + nt * 16 + col];
    #pragma unroll
    for (int r = 0; r < 8; ++r)
      out[(size_t)(row0 + base8 + r) * E_ + n0 + nt * 16 + col] = c[nt][r] + bias;
  }
}

extern "C" void kernel_launch(void* const* d_in, const int* in_sizes, int n_in,
                              void* d_out, int out_size, void* d_ws, size_t ws_size,
                              hipStream_t stream) {
  const float* x  = (const float*)d_in[0];
  const float* Wq = (const float*)d_in[1];
  const float* Wk = (const float*)d_in[2];
  const float* Wv = (const float*)d_in[3];
  const float* Wp = (const float*)d_in[4];
  const float* bp = (const float*)d_in[5];
  float* out = (float*)d_out;

  char* ws = (char*)d_ws;
  size_t off = 0;
  auto carve = [&](size_t bytes) -> void* {
    void* p = ws + off;
    off += (bytes + 255) & ~(size_t)255;
    return p;
  };
  _Float16* qh    = (_Float16*)carve((size_t)BT_ * HS_ * 2);  // 1 MB
  _Float16* kh    = (_Float16*)carve((size_t)BT_ * HS_ * 2);  // 1 MB
  _Float16* vt    = (_Float16*)carve((size_t)BT_ * HS_ * 2);  // 1 MB (transposed per batch)
  _Float16* headh = (_Float16*)carve((size_t)BT_ * HS_ * 2);  // 1 MB
  _Float16* WqT   = (_Float16*)carve((size_t)HS_ * E_ * 2);
  _Float16* WkT   = (_Float16*)carve((size_t)HS_ * E_ * 2);
  _Float16* WvT   = (_Float16*)carve((size_t)HS_ * E_ * 2);
  _Float16* WpT   = (_Float16*)carve((size_t)E_ * HS_ * 2);
  (void)in_sizes; (void)n_in; (void)out_size; (void)ws_size;

  prep_weights<<<256, 256, 0, stream>>>(Wq, Wk, Wv, Wp, WqT, WkT, WvT, WpT);
  qkv_wmma<<<64, 256, 0, stream>>>(x, WqT, WkT, WvT, qh, kh, vt);
  attn_wmma<<<128, 128, 0, stream>>>(qh, kh, vt, headh);
  proj_wmma<<<1024, 256, 0, stream>>>(headh, WpT, bp, out);
}
